// MeshGNNDecoder_69501160783924
// MI455X (gfx1250) — matrix operand order repaired and checked by reading, not compiled, not run on hardware
//
#include <hip/hip_runtime.h>
#include <math.h>

typedef __attribute__((ext_vector_type(16))) _Float16 v16h;
typedef __attribute__((ext_vector_type(8)))  _Float16 v8h;
typedef __attribute__((ext_vector_type(8)))  float    v8f;

union Frag16 { v16h f; v8h h[2]; };

__device__ __forceinline__ float gelu_f(float x) {
    return 0.5f * x * (1.0f + erff(x * 0.70710678118654752f));
}

// monotonic-bit-pattern float atomic max (init must be -inf)
__device__ __forceinline__ void atomicMaxF(float* addr, float val) {
    if (val >= 0.0f) atomicMax((int*)addr, __float_as_int(val));
    else             atomicMin((unsigned int*)addr, __float_as_uint(val));
}

__global__ void fill_kernel(float* __restrict__ p, float v, long long cnt) {
    long long i = (long long)blockIdx.x * blockDim.x + threadIdx.x;
    long long stride = (long long)gridDim.x * blockDim.x;
    for (; i < cnt; i += stride) p[i] = v;
}

// ---------------------------------------------------------------------------
// WMMA GEMM: out[M,N] = f(A[M,128] @ B[128,N] + bias + bias2?), f16 in, f32 acc
// block = 256 threads (8 waves), tile 128(M) x 64(N), full K=128 staged in LDS
// ---------------------------------------------------------------------------
#define LSTR 136   // 128 + 8 halves pad -> 272B row stride (16B aligned)

__global__ __launch_bounds__(256)
void gemm128_wmma(const float* __restrict__ A, const float* __restrict__ B,
                  const float* __restrict__ bias, const float* __restrict__ bias2,
                  float* __restrict__ out, int M, int N, int do_gelu) {
    __shared__ _Float16 As[128 * LSTR];   // A tile, row-major [row][k]
    __shared__ _Float16 Bt[64 * LSTR];    // B chunk transposed [n][k]
    const int tid = threadIdx.x;
    const int m0 = blockIdx.x * 128;
    const int n0 = blockIdx.y * 64;

    // stage A: 128 rows x 128 K (f32 -> f16), zero-pad past M
    for (int idx = tid; idx < 128 * 32; idx += 256) {
        int r = idx >> 5, c4 = (idx & 31) << 2;
        int gr = m0 + r;
        float4 v = make_float4(0.f, 0.f, 0.f, 0.f);
        if (gr < M) v = *(const float4*)(A + (size_t)gr * 128 + c4);
        _Float16* d = As + r * LSTR + c4;
        d[0] = (_Float16)v.x; d[1] = (_Float16)v.y;
        d[2] = (_Float16)v.z; d[3] = (_Float16)v.w;
    }
    // stage B chunk transposed: Bt[n][k] = B[k][n0+n]
    for (int idx = tid; idx < 128 * 64; idx += 256) {
        int k = idx >> 6, nn = idx & 63;
        Bt[nn * LSTR + k] = (_Float16)B[(size_t)k * N + (n0 + nn)];
    }
    __syncthreads();

    const int lan = tid & 31;
    const int w   = tid >> 5;           // wave -> rows [16w, 16w+16)
    const int hi  = lan >> 4;           // lane group 0/1
    const int nl  = lan & 15;
    const int arow = w * 16 + nl;

    // hoist the 4 A fragments (16x32 f16 layout, CDNA5 ISA 7.12.2)
    Frag16 a[4];
#pragma unroll
    for (int kc = 0; kc < 4; ++kc) {
        int k0 = kc * 32 + hi * 8;
        a[kc].h[0] = *(const v8h*)(As + arow * LSTR + k0);        // K k0..k0+7
        a[kc].h[1] = *(const v8h*)(As + arow * LSTR + k0 + 16);   // K k0+16..+23
    }
    const int rbase = hi * 8;

#pragma unroll
    for (int t = 0; t < 4; ++t) {       // 4 x 16-wide N tiles
        v8f acc = {};
        const int nn = t * 16 + nl;
#pragma unroll
        for (int kc = 0; kc < 4; ++kc) {
            Frag16 b;                   // 32x16 f16 B fragment: K k0..k0+7, k0+16..k0+23
            int k0 = kc * 32 + hi * 8;
            b.h[0] = *(const v8h*)(Bt + nn * LSTR + k0);
            b.h[1] = *(const v8h*)(Bt + nn * LSTR + k0 + 16);
            acc = __builtin_amdgcn_wmma_f32_16x16x32_f16(
                false, a[kc].f, false, b.f, (short)0, acc, false, false);
        }
        const int gcol = n0 + t * 16 + nl;
        float badd = bias[gcol] + (bias2 ? bias2[gcol] : 0.0f);
#pragma unroll
        for (int r = 0; r < 8; ++r) {   // C/D layout: M = rbase+r, N = nl
            int grow = m0 + w * 16 + rbase + r;
            if (grow < M) {
                float v = acc[r] + badd;
                if (do_gelu) v = gelu_f(v);
                out[(size_t)grow * N + gcol] = v;
            }
        }
    }
}

// ---------------------------------------------------------------------------
// small dense kernels
// ---------------------------------------------------------------------------
__global__ void node_embed1_kernel(const float* __restrict__ V, const float* __restrict__ W,
                                   const float* __restrict__ b, float* __restrict__ out, int n) {
    int idx = blockIdx.x * blockDim.x + threadIdx.x;
    if (idx >= n * 128) return;
    int row = idx >> 7, c = idx & 127;
    const float* v = V + (size_t)row * 3;
    out[idx] = gelu_f(v[0] * W[c] + v[1] * W[128 + c] + v[2] * W[256 + c] + b[c]);
}

__global__ void style_kernel(const float* __restrict__ z, const float* __restrict__ W1,
                             const float* __restrict__ b1, const float* __restrict__ W2,
                             const float* __restrict__ b2, float* __restrict__ out) {
    __shared__ float hid[128];
    int c = threadIdx.x;
    float s = b1[c];
    for (int k = 0; k < 256; ++k) s += z[k] * W1[k * 128 + c];
    hid[c] = gelu_f(s);
    __syncthreads();
    float s2 = b2[c];
    for (int k = 0; k < 128; ++k) s2 += hid[k] * W2[k * 128 + c];
    out[c] = s2;
}

__global__ void head2_kernel(const float* __restrict__ h1, const float* __restrict__ W,
                             const float* __restrict__ b, float* __restrict__ out, int n) {
    int idx = blockIdx.x * blockDim.x + threadIdx.x;
    if (idx >= n * 3) return;
    int row = idx / 3, j = idx - row * 3;
    const float* hr = h1 + (size_t)row * 64;
    float s = b[j];
    for (int k = 0; k < 64; ++k) s += hr[k] * W[k * 3 + j];
    out[idx] = s;
}

// ---------------------------------------------------------------------------
// GATv2 edge phase (wave-per-edge, self loops synthesized for e >= E)
// ---------------------------------------------------------------------------
__global__ __launch_bounds__(256)
void edge_logits_kernel(const int* __restrict__ src, const int* __restrict__ dst,
                        int E, int Et, const float* __restrict__ xl,
                        const float* __restrict__ xr, const float* __restrict__ att,
                        int H, float* __restrict__ lg, float* __restrict__ mbuf) {
    int wid = blockIdx.x * 8 + (threadIdx.x >> 5);
    if (wid >= Et) return;
    int lan = threadIdx.x & 31;
    int s = wid < E ? src[wid] : wid - E;
    int d = wid < E ? dst[wid] : wid - E;
    for (int h = 0; h < H; ++h) {
        float4 a  = *(const float4*)(xl + ((size_t)s * H + h) * 128 + lan * 4);
        float4 bq = *(const float4*)(xr + ((size_t)d * H + h) * 128 + lan * 4);
        float4 at = *(const float4*)(att + (size_t)h * 128 + lan * 4);
        float p = 0.f, t;
        t = a.x + bq.x; t = t > 0.f ? t : 0.2f * t; p += t * at.x;
        t = a.y + bq.y; t = t > 0.f ? t : 0.2f * t; p += t * at.y;
        t = a.z + bq.z; t = t > 0.f ? t : 0.2f * t; p += t * at.z;
        t = a.w + bq.w; t = t > 0.f ? t : 0.2f * t; p += t * at.w;
        for (int o = 16; o > 0; o >>= 1) p += __shfl_xor(p, o, 32);
        if (lan == 0) {
            lg[(size_t)wid * H + h] = p;
            atomicMaxF(&mbuf[(size_t)d * H + h], p);
        }
    }
}

__global__ void edge_exp_kernel(const int* __restrict__ dst, int E, int Et, int H,
                                float* __restrict__ lg, const float* __restrict__ mbuf,
                                float* __restrict__ den) {
    long long idx = (long long)blockIdx.x * blockDim.x + threadIdx.x;
    if (idx >= (long long)Et * H) return;
    int e = (int)(idx / H), h = (int)(idx - (long long)e * H);
    int d = e < E ? dst[e] : e - E;
    float mm = mbuf[(size_t)d * H + h];
    if (!isfinite(mm)) mm = 0.f;
    float p = expf(lg[idx] - mm);
    lg[idx] = p;
    atomicAdd(&den[(size_t)d * H + h], p);
}

__global__ __launch_bounds__(256)
void edge_scatter_kernel(const int* __restrict__ src, const int* __restrict__ dst,
                         int E, int Et, const float* __restrict__ xl,
                         const float* __restrict__ lg, const float* __restrict__ den,
                         int H, float* __restrict__ acc) {
    int wid = blockIdx.x * 8 + (threadIdx.x >> 5);
    if (wid >= Et) return;
    int lan = threadIdx.x & 31;
    int s = wid < E ? src[wid] : wid - E;
    int d = wid < E ? dst[wid] : wid - E;
    for (int h = 0; h < H; ++h) {
        float alpha = lg[(size_t)wid * H + h] / (den[(size_t)d * H + h] + 1e-16f);
        float4 a = *(const float4*)(xl + ((size_t)s * H + h) * 128 + lan * 4);
        float* o = acc + ((size_t)d * H + h) * 128 + lan * 4;
        atomicAdd(o + 0, alpha * a.x);
        atomicAdd(o + 1, alpha * a.y);
        atomicAdd(o + 2, alpha * a.z);
        atomicAdd(o + 3, alpha * a.w);
    }
}

__global__ void node_mean_kernel(const float* __restrict__ acc, const float* __restrict__ bo,
                                 int H, float* __restrict__ xn, int n) {
    int idx = blockIdx.x * blockDim.x + threadIdx.x;
    if (idx >= n * 128) return;
    int row = idx >> 7, c = idx & 127;
    float v;
    if (H == 2) v = 0.5f * (acc[(size_t)row * 256 + c] + acc[(size_t)row * 256 + 128 + c]);
    else        v = acc[(size_t)row * 128 + c];
    xn[idx] = v + bo[c];
}

// ---------------------------------------------------------------------------
// GraphNorm (single graph): column mean/var reductions + normalize + GELU
// ---------------------------------------------------------------------------
__global__ void gn_colsum_kernel(const float* __restrict__ x, float* __restrict__ mu,
                                 int n, int rpb) {
    int c = threadIdx.x;
    int r0 = blockIdx.x * rpb;
    int r1 = r0 + rpb; if (r1 > n) r1 = n;
    if (r0 >= r1) return;
    float s = 0.f;
    for (int r = r0; r < r1; ++r) s += x[(size_t)r * 128 + c];
    atomicAdd(&mu[c], s);
}

__global__ void gn_var_kernel(const float* __restrict__ x, const float* __restrict__ mu,
                              float* __restrict__ var, const float* __restrict__ ms,
                              int n, int rpb) {
    int c = threadIdx.x;
    int r0 = blockIdx.x * rpb;
    int r1 = r0 + rpb; if (r1 > n) r1 = n;
    if (r0 >= r1) return;
    float m = ms[c] * (mu[c] / (float)n);
    float s = 0.f;
    for (int r = r0; r < r1; ++r) {
        float d = x[(size_t)r * 128 + c] - m;
        s += d * d;
    }
    atomicAdd(&var[c], s);
}

__global__ void gn_norm_kernel(const float* __restrict__ x, const float* __restrict__ mu,
                               const float* __restrict__ var, const float* __restrict__ ms,
                               const float* __restrict__ w, const float* __restrict__ b,
                               float* __restrict__ out, int n) {
    int idx = blockIdx.x * blockDim.x + threadIdx.x;
    if (idx >= n * 128) return;
    int c = idx & 127;
    float invn = 1.0f / (float)n;
    float o = x[idx] - ms[c] * (mu[c] * invn);
    float y = w[c] * o * rsqrtf(var[c] * invn + 1e-5f) + b[c];
    out[idx] = gelu_f(y);
}

// ---------------------------------------------------------------------------
extern "C" void kernel_launch(void* const* d_in, const int* in_sizes, int n_in,
                              void* d_out, int out_size, void* d_ws, size_t ws_size,
                              hipStream_t stream) {
    (void)n_in; (void)out_size; (void)ws_size;
    const float* z_g   = (const float*)d_in[0];
    const int*   ei    = (const int*)d_in[1];
    const float* verts = (const float*)d_in[2];
    const float* W_ne1 = (const float*)d_in[3];
    const float* b_ne1 = (const float*)d_in[4];
    const float* W_ne2 = (const float*)d_in[5];
    const float* b_ne2 = (const float*)d_in[6];
    const float* W_s1  = (const float*)d_in[7];
    const float* b_s1  = (const float*)d_in[8];
    const float* W_s2  = (const float*)d_in[9];
    const float* b_s2  = (const float*)d_in[10];
    const float* W_h1  = (const float*)d_in[38];
    const float* b_h1  = (const float*)d_in[39];
    const float* W_h2  = (const float*)d_in[40];
    const float* b_h2  = (const float*)d_in[41];

    const int n  = in_sizes[2] / 3;
    const int E  = in_sizes[1] / 2;
    const int Et = E + n;
    const int* srcp = ei;
    const int* dstp = ei + E;

    // workspace layout (floats)
    float* ws = (float*)d_ws;
    size_t off = 0;
    float* xbuf  = ws + off; off += (size_t)n * 128;  // current features x
    float* xl    = ws + off; off += (size_t)n * 256;  // also reused for x0 / h1
    float* xr    = ws + off; off += (size_t)n * 256;
    float* acc   = ws + off; off += (size_t)n * 256;
    float* xn    = ws + off; off += (size_t)n * 128;
    float* lg    = ws + off; off += (size_t)Et * 2;
    float* mbuf  = ws + off; off += (size_t)n * 2;
    float* den   = ws + off; off += (size_t)n * 2;
    float* stats = ws + off; off += 256;              // mu[128], var[128]
    float* stylv = ws + off; off += 128;

    auto cdiv = [](long long a, long long b) { return (int)((a + b - 1) / b); };
    const int gmx = cdiv(n, 128);
    const int GNB = 512;
    const int rpb = cdiv(n, GNB);

    // node embedding: x0 (in xl) = gelu(V @ W_ne1 + b)
    node_embed1_kernel<<<cdiv((long long)n * 128, 256), 256, 0, stream>>>(
        verts, W_ne1, b_ne1, xl, n);
    // style vector (128)
    style_kernel<<<1, 128, 0, stream>>>(z_g, W_s1, b_s1, W_s2, b_s2, stylv);
    // x = x0 @ W_ne2 + b_ne2 + style
    gemm128_wmma<<<dim3(gmx, 1), 256, 0, stream>>>(xl, W_ne2, b_ne2, stylv, xbuf, n, 128, 0);

    for (int L = 0; L < 3; ++L) {
        const int base = 11 + 9 * L;
        const float* Wl  = (const float*)d_in[base + 0];
        const float* bl  = (const float*)d_in[base + 1];
        const float* Wr  = (const float*)d_in[base + 2];
        const float* br  = (const float*)d_in[base + 3];
        const float* att = (const float*)d_in[base + 4];
        const float* bo  = (const float*)d_in[base + 5];
        const float* gw  = (const float*)d_in[base + 6];
        const float* gb  = (const float*)d_in[base + 7];
        const float* gms = (const float*)d_in[base + 8];
        const int H  = (L < 2) ? 2 : 1;
        const int NN = H * 128;

        gemm128_wmma<<<dim3(gmx, NN / 64), 256, 0, stream>>>(xbuf, Wl, bl, nullptr, xl, n, NN, 0);
        gemm128_wmma<<<dim3(gmx, NN / 64), 256, 0, stream>>>(xbuf, Wr, br, nullptr, xr, n, NN, 0);

        fill_kernel<<<256, 256, 0, stream>>>(mbuf, -INFINITY, (long long)n * H);
        fill_kernel<<<256, 256, 0, stream>>>(den, 0.0f, (long long)n * H);
        fill_kernel<<<512, 256, 0, stream>>>(acc, 0.0f, (long long)n * NN);

        edge_logits_kernel<<<cdiv(Et, 8), 256, 0, stream>>>(srcp, dstp, E, Et, xl, xr, att, H, lg, mbuf);
        edge_exp_kernel<<<cdiv((long long)Et * H, 256), 256, 0, stream>>>(dstp, E, Et, H, lg, mbuf, den);
        edge_scatter_kernel<<<cdiv(Et, 8), 256, 0, stream>>>(srcp, dstp, E, Et, xl, lg, den, H, acc);
        node_mean_kernel<<<cdiv((long long)n * 128, 256), 256, 0, stream>>>(acc, bo, H, xn, n);

        fill_kernel<<<1, 256, 0, stream>>>(stats, 0.0f, 256);
        gn_colsum_kernel<<<GNB, 128, 0, stream>>>(xn, stats, n, rpb);
        gn_var_kernel<<<GNB, 128, 0, stream>>>(xn, stats, stats + 128, gms, n, rpb);
        gn_norm_kernel<<<cdiv((long long)n * 128, 256), 256, 0, stream>>>(
            xn, stats, stats + 128, gms, gw, gb, xbuf, n);
    }

    // head: h1 (in xl) = gelu(x @ W_h1 + b_h1); out = h1 @ W_h2 + b_h2
    gemm128_wmma<<<dim3(gmx, 1), 256, 0, stream>>>(xbuf, W_h1, b_h1, nullptr, xl, n, 64, 1);
    head2_kernel<<<cdiv((long long)n * 3, 256), 256, 0, stream>>>(xl, W_h2, b_h2, (float*)d_out, n);
}
